// HyenaOperator_37864431681934
// MI455X (gfx1250) — matrix operand order repaired and checked by reading, not compiled, not run on hardware
//
#include <hip/hip_runtime.h>
#include <cstdint>
#include <cstddef>

typedef _Float16 half_t;
typedef __attribute__((ext_vector_type(16))) _Float16 v16h;
typedef __attribute__((ext_vector_type(8)))  float    v8f;
typedef __attribute__((ext_vector_type(4)))  float    vf4;
typedef __attribute__((ext_vector_type(4)))  int      v4i;

#define BD 768      // model dim D
#define LL 2048     // sequence length L
#define BB 2        // batch
#define FF 64       // filter basis F
#define ML (BB*LL)  // 4096 rows total

// ---- CDNA5 async global->LDS copy (ASYNCcnt-tracked), with fallback --------
#if defined(__has_builtin)
#if __has_builtin(__builtin_amdgcn_global_load_async_to_lds_b128)
#define USE_ASYNC_LDS 1
#endif
#endif

#define AS_GLOBAL __attribute__((address_space(1)))
#define AS_LOCAL  __attribute__((address_space(3)))

__device__ __forceinline__ void async_cp16(const void* g, void* l) {
#ifdef USE_ASYNC_LDS
  __builtin_amdgcn_global_load_async_to_lds_b128(
      (AS_GLOBAL v4i*)g, (AS_LOCAL v4i*)l, 0, 0);
#endif
}

__device__ __forceinline__ void wait_async0() {
#if defined(__has_builtin) && __has_builtin(__builtin_amdgcn_s_wait_asynccnt)
  __builtin_amdgcn_s_wait_asynccnt(0);
#else
  asm volatile("s_wait_asynccnt 0" ::: "memory");
#endif
}

struct HalfPair { vf4 lo, hi; };
__device__ __forceinline__ v16h make_frag(vf4 lo, vf4 hi) {
  HalfPair p{lo, hi};
  return __builtin_bit_cast(v16h, p);
}

// ---------------------------------------------------------------------------
// f32 -> f16 elementwise convert
// ---------------------------------------------------------------------------
__global__ __launch_bounds__(256) void k_cvt_half(const float* __restrict__ in,
                                                  half_t* __restrict__ outp) {
  size_t idx = (size_t)blockIdx.x * 256 + threadIdx.x;
  outp[idx] = (half_t)in[idx];
}

// ---------------------------------------------------------------------------
// transpose + convert: in (R,C) f32 row-major -> out (C,R) f16
// ---------------------------------------------------------------------------
__global__ __launch_bounds__(256) void k_transpose_cvt(const float* __restrict__ in,
                                                       half_t* __restrict__ outp,
                                                       int R, int C) {
  __shared__ float tile[32][33];
  int tx = threadIdx.x, ty = threadIdx.y;
  int c0 = blockIdx.x * 32, r0 = blockIdx.y * 32;
#pragma unroll
  for (int j = 0; j < 32; j += 8)
    tile[ty + j][tx] = in[(size_t)(r0 + ty + j) * C + c0 + tx];
  __syncthreads();
#pragma unroll
  for (int j = 0; j < 32; j += 8)
    outp[(size_t)(c0 + ty + j) * R + r0 + tx] = (half_t)tile[tx][ty + j];
}

// ---------------------------------------------------------------------------
// batched f32 transpose: in (b,R,C) -> out (b,C,R)
// ---------------------------------------------------------------------------
__global__ __launch_bounds__(256) void k_transpose_f32(const float* __restrict__ in,
                                                       float* __restrict__ outp,
                                                       int R, int C) {
  __shared__ float tile[32][33];
  int b = blockIdx.z;
  const float* ip = in + (size_t)b * R * C;
  float* op = outp + (size_t)b * R * C;
  int tx = threadIdx.x, ty = threadIdx.y;
  int c0 = blockIdx.x * 32, r0 = blockIdx.y * 32;
#pragma unroll
  for (int j = 0; j < 32; j += 8)
    tile[ty + j][tx] = ip[(size_t)(r0 + ty + j) * C + c0 + tx];
  __syncthreads();
#pragma unroll
  for (int j = 0; j < 32; j += 8)
    op[(size_t)(c0 + ty + j) * R + r0 + tx] = tile[tx][ty + j];
}

// ---------------------------------------------------------------------------
// short depthwise conv (k=3, pad=1)
// ---------------------------------------------------------------------------
__global__ __launch_bounds__(256) void k_short(const float* __restrict__ x,
                                               const float* __restrict__ w3,
                                               const float* __restrict__ bsh,
                                               float* __restrict__ outp) {
  size_t idx = (size_t)blockIdx.x * 256 + threadIdx.x;
  int d = (int)(idx % BD);
  size_t row = idx / BD;   // b*L + i
  int i = (int)(row % LL);
  float acc = bsh[d] + x[idx] * w3[d * 3 + 1];
  if (i > 0)      acc += x[idx - BD] * w3[d * 3 + 0];
  if (i < LL - 1) acc += x[idx + BD] * w3[d * 3 + 2];
  outp[idx] = acc;
}

// ---------------------------------------------------------------------------
// sinusoidal positional basis: basis[k,f] = sin(k*f/F)
// ---------------------------------------------------------------------------
__global__ __launch_bounds__(256) void k_basis(float* __restrict__ basis) {
  int idx = blockIdx.x * 256 + threadIdx.x;
  int k = idx >> 6, f = idx & 63;
  basis[idx] = sinf((float)k * (float)f * (1.0f / (float)FF));
}

// ---------------------------------------------------------------------------
// implicit filter: filt[d,k] = b_filt[d] + sum_f basis[k,f]*w_filt[f,d]
// ---------------------------------------------------------------------------
__global__ __launch_bounds__(256) void k_filt(const float* __restrict__ basis,
                                              const float* __restrict__ w_filt,
                                              const float* __restrict__ b_filt,
                                              float* __restrict__ filt) {
  __shared__ float wf[FF];
  int d = blockIdx.y;
  int k = blockIdx.x * 256 + threadIdx.x;
  if (threadIdx.x < FF) wf[threadIdx.x] = w_filt[(size_t)threadIdx.x * BD + d];
  __syncthreads();
  float acc = b_filt[d];
  const float* brow = basis + (size_t)k * FF;
#pragma unroll 8
  for (int f = 0; f < FF; ++f) acc += brow[f] * wf[f];
  filt[(size_t)d * LL + k] = acc;
}

// ---------------------------------------------------------------------------
// gating: g = silu(z) * v1 * v2  (in-place over z)
// ---------------------------------------------------------------------------
__global__ __launch_bounds__(256) void k_gate(float* __restrict__ z,
                                              const float* __restrict__ v1,
                                              const float* __restrict__ v2) {
  size_t idx = (size_t)blockIdx.x * 256 + threadIdx.x;
  float zv = z[idx];
  float s = zv / (1.0f + __expf(-zv));
  z[idx] = s * v1[idx] * v2[idx];
}

// ---------------------------------------------------------------------------
// long depthwise conv, kernel length L, pad L/2, truncated to L.
// One WG per (b,d): padded signal (16KB) + filter (8KB) resident in LDS.
// LDS fill uses CDNA5 async global->LDS copies (ASYNCcnt) when available.
// ---------------------------------------------------------------------------
__global__ __launch_bounds__(256) void k_conv(const float* __restrict__ xprojT,
                                              const float* __restrict__ filt,
                                              float* __restrict__ convT) {
  __shared__ __align__(16) float xp[2 * LL];
  __shared__ __align__(16) float fl[LL];
  int d = blockIdx.x, b = blockIdx.y, tid = threadIdx.x;
  const float* xrow = xprojT + ((size_t)b * BD + d) * LL;
  const float* frow = filt + (size_t)d * LL;
#ifdef USE_ASYNC_LDS
  vf4 z4 = {0.f, 0.f, 0.f, 0.f};
  *(vf4*)&xp[tid * 4] = z4;            // left pad  [0,1024)
  *(vf4*)&xp[3072 + tid * 4] = z4;     // right pad [3072,4096)
#pragma unroll
  for (int it = 0; it < 2; ++it) {
    int c = tid + it * 256;            // 512 x 16B chunks of signal
    async_cp16(xrow + c * 4, &xp[1024 + c * 4]);
    async_cp16(frow + c * 4, &fl[c * 4]);
  }
  wait_async0();
  __syncthreads();
#else
  for (int j = tid; j < 2 * LL; j += 256)
    xp[j] = (j >= LL / 2 && j < LL / 2 + LL) ? xrow[j - LL / 2] : 0.0f;
  for (int j = tid; j < LL; j += 256) fl[j] = frow[j];
  __syncthreads();
#endif

  float acc[8] = {0.f, 0.f, 0.f, 0.f, 0.f, 0.f, 0.f, 0.f};
  const int i0 = tid * 4;
  for (int k = 0; k < LL; k += 4) {
    vf4 f4 = *(const vf4*)&fl[k];
    vf4 xa = *(const vf4*)&xp[i0 + k];
    vf4 xb = *(const vf4*)&xp[i0 + k + 4];
    vf4 ya = *(const vf4*)&xp[i0 + 1024 + k];
    vf4 yb = *(const vf4*)&xp[i0 + 1024 + k + 4];
#pragma unroll
    for (int c = 0; c < 4; ++c) {
#pragma unroll
      for (int q = 0; q < 4; ++q) {
        float xv = (c + q < 4) ? xa[c + q] : xb[c + q - 4];
        float yv = (c + q < 4) ? ya[c + q] : yb[c + q - 4];
        acc[c]     += xv * f4[q];
        acc[4 + c] += yv * f4[q];
      }
    }
  }
  float* orow = convT + ((size_t)b * BD + d) * LL;
  vf4 o0 = {acc[0], acc[1], acc[2], acc[3]};
  vf4 o1 = {acc[4], acc[5], acc[6], acc[7]};
  *(vf4*)&orow[i0] = o0;
  *(vf4*)&orow[i0 + 1024] = o1;
}

// ---------------------------------------------------------------------------
// fuse: y[b,i,d] = (half)(convT[b,d,i] * g[b,i,d] + x_short[b,i,d])
// ---------------------------------------------------------------------------
__global__ __launch_bounds__(256) void k_gather_y(const float* __restrict__ convT,
                                                  const float* __restrict__ g,
                                                  const float* __restrict__ xs,
                                                  half_t* __restrict__ y) {
  __shared__ float tile[32][33];
  int b = blockIdx.z;
  const float* cp = convT + (size_t)b * BD * LL;
  int tx = threadIdx.x, ty = threadIdx.y;
  int i0 = blockIdx.x * 32, d0 = blockIdx.y * 32;
#pragma unroll
  for (int j = 0; j < 32; j += 8)
    tile[ty + j][tx] = cp[(size_t)(d0 + ty + j) * LL + i0 + tx];
  __syncthreads();
#pragma unroll
  for (int j = 0; j < 32; j += 8) {
    int i = i0 + ty + j, d = d0 + tx;
    size_t idx = ((size_t)b * LL + i) * BD + d;
    y[idx] = (half_t)(tile[tx][ty + j] * g[idx] + xs[idx]);
  }
}

// ---------------------------------------------------------------------------
// WMMA GEMM: C(MxN) = A(MxK,f16) * Bt(NxK,f16)^T + bias, f32 accumulate.
// WG = 256 threads = 8 waves arranged 2(M) x 4(N); wave tile 32x64
// (2x4 accumulators of v_wmma_f32_16x16x32_f16); WG tile 64x256, K step 32.
// Double-buffered LDS; tiles staged with async global->LDS copies so the
// k+1 copy overlaps the k-tile's 8 WMMAs per wave.
// split==1: scatter columns into 4 contiguous (M x 768) blocks (proj split).
// ---------------------------------------------------------------------------
#define GTM 64
#define GTN 256
#define GTK 32

__global__ __launch_bounds__(256) void k_gemm_wmma(const half_t* __restrict__ A,
                                                   const half_t* __restrict__ Bt,
                                                   const float* __restrict__ bias,
                                                   float* __restrict__ outp,
                                                   int M, int N, int K, int split) {
  __shared__ __align__(16) half_t As[2][GTM * GTK];   // 2 x 4KB
  __shared__ __align__(16) half_t Bs[2][GTN * GTK];   // 2 x 16KB
  const int tid = threadIdx.x;
  const int m0 = blockIdx.y * GTM;
  const int n0 = blockIdx.x * GTN;
  const int lane = tid & 31, wv = tid >> 5;
  const int wm = wv >> 2, wn = wv & 3;
  const int l16 = lane & 15, hi = lane >> 4;
  const int arow = tid >> 2, ach = tid & 3;       // A: 64 rows x 4 chunks

  const half_t* gA = A + (size_t)(m0 + arow) * K + ach * 8;

  v8f acc[2][4] = {};

#ifdef USE_ASYNC_LDS
  // ---- async path: issue copies for tile kk into buffer buf -----------------
  auto issue_tile = [&](int buf, int kk) {
    async_cp16(gA + kk, &As[buf][arow * 32 + ach * 8]);
#pragma unroll
    for (int it = 0; it < 4; ++it) {
      int idx = tid + it * 256;                   // 256 rows x 4 chunks
      int brow = idx >> 2, bch = idx & 3;
      async_cp16(Bt + (size_t)(n0 + brow) * K + kk + bch * 8,
                 &Bs[buf][brow * 32 + bch * 8]);
    }
  };
  issue_tile(0, 0);
  wait_async0();
  __syncthreads();
#else
  // ---- fallback: register staging ------------------------------------------
  {
    vf4 ra = *(const vf4*)(gA);
    vf4 rb[4];
#pragma unroll
    for (int it = 0; it < 4; ++it) {
      int idx = tid + it * 256;
      rb[it] = *(const vf4*)&Bt[(size_t)(n0 + (idx >> 2)) * K + (idx & 3) * 8];
    }
    *(vf4*)&As[0][arow * 32 + ach * 8] = ra;
#pragma unroll
    for (int it = 0; it < 4; ++it) {
      int idx = tid + it * 256;
      *(vf4*)&Bs[0][(idx >> 2) * 32 + (idx & 3) * 8] = rb[it];
    }
  }
  __syncthreads();
#endif

  for (int kk = 0; kk < K; kk += GTK) {
    const int cur = (kk >> 5) & 1;
    const bool has_next = (kk + GTK) < K;

#ifdef USE_ASYNC_LDS
    if (has_next) issue_tile(cur ^ 1, kk + GTK);
#else
    vf4 ra;
    vf4 rb[4];
    if (has_next) {
      ra = *(const vf4*)(gA + kk + GTK);
#pragma unroll
      for (int it = 0; it < 4; ++it) {
        int idx = tid + it * 256;
        rb[it] = *(const vf4*)&Bt[(size_t)(n0 + (idx >> 2)) * K + kk + GTK + (idx & 3) * 8];
      }
    }
#endif

    // A fragment (ISA 16-bit A 16x32 layout): elem e -> K = (e>>3)*16 + hi*8 + (e&7)
    v16h af[2], bf[4];
#pragma unroll
    for (int mt = 0; mt < 2; ++mt) {
      const half_t* p = &As[cur][(wm * 32 + mt * 16 + l16) * 32 + hi * 8];
      af[mt] = make_frag(*(const vf4*)p, *(const vf4*)(p + 16));
    }
    // B fragment: column n = l16, elem e -> K = (e>>3)*16 + hi*8 + (e&7)
#pragma unroll
    for (int nt = 0; nt < 4; ++nt) {
      const half_t* p = &Bs[cur][(wn * 64 + nt * 16 + l16) * 32 + hi * 8];
      bf[nt] = make_frag(*(const vf4*)p, *(const vf4*)(p + 16));
    }
#pragma unroll
    for (int mt = 0; mt < 2; ++mt)
#pragma unroll
      for (int nt = 0; nt < 4; ++nt)
        acc[mt][nt] = __builtin_amdgcn_wmma_f32_16x16x32_f16(
            false, af[mt], false, bf[nt], (short)0, acc[mt][nt], false, false);

#ifdef USE_ASYNC_LDS
    wait_async0();
    __syncthreads();
#else
    if (has_next) {
      *(vf4*)&As[cur ^ 1][arow * 32 + ach * 8] = ra;
#pragma unroll
      for (int it = 0; it < 4; ++it) {
        int idx = tid + it * 256;
        *(vf4*)&Bs[cur ^ 1][(idx >> 2) * 32 + (idx & 3) * 8] = rb[it];
      }
    }
    __syncthreads();
#endif
  }

  // epilogue: C/D layout — vgpr r holds row (hi*8 + r), column l16
#pragma unroll
  for (int nt = 0; nt < 4; ++nt) {
    int col = n0 + wn * 64 + nt * 16 + l16;
    float bv = bias[col];
    int chunk = col / BD, dcol = col - chunk * BD;
#pragma unroll
    for (int mt = 0; mt < 2; ++mt) {
#pragma unroll
      for (int r = 0; r < 8; ++r) {
        int row = m0 + wm * 32 + mt * 16 + hi * 8 + r;
        float v = acc[mt][nt][r] + bv;
        if (split) outp[((size_t)chunk * ML + row) * BD + dcol] = v;
        else       outp[(size_t)row * N + col] = v;
      }
    }
  }
}

// ---------------------------------------------------------------------------
extern "C" void kernel_launch(void* const* d_in, const int* in_sizes, int n_in,
                              void* d_out, int out_size, void* d_ws, size_t ws_size,
                              hipStream_t stream) {
  const float* x       = (const float*)d_in[0];
  const float* w_in    = (const float*)d_in[1];
  const float* b_in    = (const float*)d_in[2];
  const float* w_filt  = (const float*)d_in[3];
  const float* b_filt  = (const float*)d_in[4];
  const float* w_short = (const float*)d_in[5];
  const float* b_short = (const float*)d_in[6];
  const float* w_out   = (const float*)d_in[7];
  const float* b_out   = (const float*)d_in[8];
  float* outp = (float*)d_out;

  char* ws = (char*)d_ws;
  size_t off = 0;
  auto alloc = [&](size_t bytes) -> void* {
    off = (off + 255) & ~(size_t)255;
    void* p = ws + off;
    off += bytes;
    return p;
  };
  half_t* hx     = (half_t*)alloc((size_t)ML * BD * sizeof(half_t));
  half_t* hwinT  = (half_t*)alloc((size_t)4 * BD * BD * sizeof(half_t));
  half_t* hwoutT = (half_t*)alloc((size_t)BD * BD * sizeof(half_t));
  float*  proj   = (float*)alloc((size_t)4 * ML * BD * sizeof(float));
  float*  xshort = (float*)alloc((size_t)ML * BD * sizeof(float));
  float*  basis  = (float*)alloc((size_t)LL * FF * sizeof(float));
  float*  filt   = (float*)alloc((size_t)BD * LL * sizeof(float));
  // block reuse after gating consumes z/v1/v2:
  float*  zbuf   = proj + (size_t)1 * ML * BD;   // z -> g (in place)
  float*  v1buf  = proj + (size_t)2 * ML * BD;   // -> xprojT
  float*  v2buf  = proj + (size_t)3 * ML * BD;   // -> convT
  float*  xprojT = v1buf;
  float*  convT  = v2buf;
  half_t* hy     = hx;                           // reuse x_f16 region for y_f16

  dim3 tb(32, 8);

  // 1) precision conversions / weight pre-transposes (K-contiguous f16)
  k_cvt_half<<<(ML * BD) / 256, 256, 0, stream>>>(x, hx);
  k_transpose_cvt<<<dim3(4 * BD / 32, BD / 32), tb, 0, stream>>>(w_in, hwinT, BD, 4 * BD);
  k_transpose_cvt<<<dim3(BD / 32, BD / 32), tb, 0, stream>>>(w_out, hwoutT, BD, BD);

  // 2) cheap branches: short conv, basis, implicit filter
  k_short<<<(ML * BD) / 256, 256, 0, stream>>>(x, w_short, b_short, xshort);
  k_basis<<<(LL * FF) / 256, 256, 0, stream>>>(basis);
  k_filt<<<dim3(LL / 256, BD), 256, 0, stream>>>(basis, w_filt, b_filt, filt);

  // 3) GEMM 1: proj = x @ w_in + b_in, scattered into 4 (M x D) blocks
  k_gemm_wmma<<<dim3(4 * BD / GTN, ML / GTM), 256, 0, stream>>>(
      hx, hwinT, b_in, proj, ML, 4 * BD, BD, /*split=*/1);

  // 4) gate g = silu(z)*v1*v2 (in place over z block)
  k_gate<<<(ML * BD) / 256, 256, 0, stream>>>(zbuf, v1buf, v2buf);

  // 5) transpose x_proj (b,L,D) -> (b,D,L), then long conv per channel
  k_transpose_f32<<<dim3(BD / 32, LL / 32, BB), tb, 0, stream>>>(proj, xprojT, LL, BD);
  k_conv<<<dim3(BD, BB), 256, 0, stream>>>(xprojT, filt, convT);

  // 6) y = conv * g + x_short, converted to f16 in (b,L,D) layout
  k_gather_y<<<dim3(LL / 32, BD / 32, BB), tb, 0, stream>>>(convT, zbuf, xshort, hy);

  // 7) GEMM 2: out = y @ w_out + b_out (f32 output)
  k_gemm_wmma<<<dim3(BD / GTN, ML / GTM), 256, 0, stream>>>(
      hy, hwoutT, b_out, outp, ML, BD, BD, /*split=*/0);
}
